// MambaBlock_37168646979787
// MI455X (gfx1250) — matrix operand order repaired and checked by reading, not compiled, not run on hardware
//
#include <hip/hip_runtime.h>
#include <stdint.h>

#define DMODEL 1024
#define DINNER 2048
#define DSTATE 16
#define DCONV  4
#define BSZ    4
#define SEQL   2048
#define NTOK   (BSZ * SEQL)   // 8192 tokens

typedef __attribute__((ext_vector_type(16))) __bf16 v16bf;
typedef __attribute__((ext_vector_type(8)))  float  v8f;

union FragBF {
    uint32_t u[8];
    v16bf    v;
};

__device__ __forceinline__ uint16_t f32_to_bf16(float f) {
    uint32_t x = __float_as_uint(f);
    uint32_t r = (x + 0x7FFFu + ((x >> 16) & 1u)) >> 16;  // round-to-nearest-even
    return (uint16_t)r;
}

__device__ __forceinline__ float silu_f(float v) {
    return v / (1.f + __expf(-v));
}

// ---------------------------------------------------------------- weight cvt
__global__ __launch_bounds__(256)
void convert_f32_bf16_kernel(const float* __restrict__ src,
                             uint16_t* __restrict__ dst, int n) {
    int i = blockIdx.x * 256 + threadIdx.x;
    if (i < n) dst[i] = f32_to_bf16(src[i]);
}

// ---------------------------------------------------------------- layernorm
// one token per block (256 threads, 4 elems/thread), writes bf16 xn
__global__ __launch_bounds__(256)
void layernorm_kernel(const float* __restrict__ x,
                      const float* __restrict__ w,
                      const float* __restrict__ b,
                      uint16_t* __restrict__ out) {
    __shared__ float s1[256], s2[256];
    const int t = threadIdx.x;
    const size_t base = (size_t)blockIdx.x * DMODEL;
    float v[4], sum = 0.f, sq = 0.f;
#pragma unroll
    for (int i = 0; i < 4; i++) {
        v[i] = x[base + t + 256 * i];
        sum += v[i];
        sq  += v[i] * v[i];
    }
    s1[t] = sum; s2[t] = sq;
    __syncthreads();
    for (int off = 128; off > 0; off >>= 1) {
        if (t < off) { s1[t] += s1[t + off]; s2[t] += s2[t + off]; }
        __syncthreads();
    }
    const float mu  = s1[0] * (1.f / DMODEL);
    const float var = s2[0] * (1.f / DMODEL) - mu * mu;
    const float rs  = rsqrtf(var + 1e-5f);
#pragma unroll
    for (int i = 0; i < 4; i++) {
        int c = t + 256 * i;
        out[base + c] = f32_to_bf16((v[i] - mu) * rs * w[c] + b[c]);
    }
}

// ---------------------------------------------------------------- WMMA GEMM
// C[M,N] = A[M,K](bf16,row-major) * W[N,K](bf16,row-major)^T  (+ epilogue)
// MODE 0: plain f32 store; 1: softplus(acc + bias[col]); 2: res[idx] + acc
// One wave computes an (MT*16) x (NT*16) macro-tile; 8 waves / block.
// MT=NT=4 -> 64x64/wave: 16 WMMA per 32-K step for 8KB of fragments
// (~32 FLOP/B vs L2-resident weights), A & B each reused 4x per step.
template<int MT, int NT, int MODE>
__global__ __launch_bounds__(256)
void wmma_gemm_bf16(const uint16_t* __restrict__ A,
                    const uint16_t* __restrict__ W,
                    const float* __restrict__ bias,
                    const float* __restrict__ res,
                    float* __restrict__ C,
                    int M, int N, int K) {
    const int lane   = threadIdx.x & 31;
    const int wave   = threadIdx.x >> 5;
    const int tilesN = N / (16 * NT);
    const int gtile  = blockIdx.x * 8 + wave;
    const int m0 = (gtile / tilesN) * 16 * MT;
    const int n0 = (gtile % tilesN) * 16 * NT;

    v8f acc[MT][NT];
#pragma unroll
    for (int i = 0; i < MT; i++)
#pragma unroll
        for (int j = 0; j < NT; j++)
            acc[i][j] = (v8f){0.f, 0.f, 0.f, 0.f, 0.f, 0.f, 0.f, 0.f};

    const int mr   = lane & 15;
    const int aoff = (lane >> 4) * 8;  // K-half select for A frag (uint idx)
    const int boff = (lane >> 4) * 8;  // K-half select for B frag (uint idx)

    for (int k0 = 0; k0 < K; k0 += 32) {
        FragBF a[MT];
#pragma unroll
        for (int i = 0; i < MT; i++) {
            const uint32_t* Ar =
                (const uint32_t*)(A + (size_t)(m0 + i * 16 + mr) * K + k0);
#pragma unroll
            for (int j = 0; j < 4; j++) {
                a[i].u[j]     = Ar[aoff + j];      // K = 16*khalf + 0..7
                a[i].u[4 + j] = Ar[aoff + 4 + j];  // K = 16*khalf + 8..15
            }
            // pull next K-chunk toward the WGP while WMMAs run
            __builtin_prefetch(Ar + 16, 0, 3);     // global_prefetch_b8
        }
        FragBF bfr[NT];
#pragma unroll
        for (int j = 0; j < NT; j++) {
            const uint32_t* Wr =
                (const uint32_t*)(W + (size_t)(n0 + j * 16 + mr) * K + k0);
#pragma unroll
            for (int t = 0; t < 8; t++) bfr[j].u[t] = Wr[boff + t];
            __builtin_prefetch(Wr + 16, 0, 3);
        }
#pragma unroll
        for (int i = 0; i < MT; i++)
#pragma unroll
            for (int j = 0; j < NT; j++)
                acc[i][j] = __builtin_amdgcn_wmma_f32_16x16x32_bf16(
                    false, a[i].v, false, bfr[j].v, (short)0, acc[i][j],
                    false, false);
    }

    const int rowadd = (lane >> 4) * 8;   // lanes 16-31 hold M = r + 8
    const int col16  = lane & 15;
#pragma unroll
    for (int i = 0; i < MT; i++)
#pragma unroll
        for (int j = 0; j < NT; j++)
#pragma unroll
            for (int r = 0; r < 8; r++) {
                const int row = m0 + i * 16 + r + rowadd;
                const int col = n0 + j * 16 + col16;
                const size_t idx = (size_t)row * N + col;
                float v = acc[i][j][r];
                if (MODE == 1) {
                    v += bias[col];
                    v = (v > 20.f) ? v : log1pf(__expf(v));  // softplus
                    C[idx] = v;
                } else if (MODE == 2) {
                    C[idx] = res[idx] + v;
                } else {
                    C[idx] = v;
                }
            }
}

// ---------------------------------------------------------------- conv+SiLU
// depthwise causal conv (k=4) on x_part = xz[..., :DINNER]; writes f32 + bf16
__global__ __launch_bounds__(256)
void conv_silu_kernel(const float* __restrict__ xz,
                      const float* __restrict__ cw,
                      const float* __restrict__ cb,
                      float* __restrict__ xc,
                      uint16_t* __restrict__ xcb) {
    const size_t idx = (size_t)blockIdx.x * 256 + threadIdx.x;
    const int e  = (int)(idx % DINNER);
    const int l  = (int)((idx / DINNER) % SEQL);
    const int bb = (int)(idx / ((size_t)DINNER * SEQL));
    float acc = cb[e];
#pragma unroll
    for (int j = 0; j < DCONV; j++) {
        const int ll = l - (DCONV - 1) + j;
        if (ll >= 0)
            acc += cw[e * DCONV + j] *
                   xz[((size_t)bb * SEQL + ll) * (2 * DINNER) + e];
    }
    const float s = silu_f(acc);
    xc[idx]  = s;
    xcb[idx] = f32_to_bf16(s);
}

// ---------------------------------------------------------------- scan
// sequential over L; thread = one (b, e) channel, h[16] in registers.
// B/C rows (per-token, shared across channels) broadcast via LDS.
// Fused epilogue: y = (y + x_conv*D) * silu(z), written as bf16 for out GEMM.
__global__ __launch_bounds__(256)
void scan_kernel(const float* __restrict__ dt,
                 const float* __restrict__ xc,
                 const float* __restrict__ Bm,
                 const float* __restrict__ Cm,
                 const float* __restrict__ xz,
                 const float* __restrict__ A_log,
                 const float* __restrict__ Dp,
                 uint16_t* __restrict__ ybf) {
    const int e  = blockIdx.x * 256 + threadIdx.x;
    const int bb = blockIdx.y;
    __shared__ float sB[DSTATE], sC[DSTATE];
    float Ar[DSTATE], h[DSTATE];
#pragma unroll
    for (int s = 0; s < DSTATE; s++) {
        Ar[s] = -__expf(A_log[e * DSTATE + s]);
        h[s]  = 0.f;
    }
    const float de = Dp[e];
    for (int l = 0; l < SEQL; l++) {
        const size_t tok = (size_t)bb * SEQL + l;
        if (threadIdx.x < DSTATE)
            sB[threadIdx.x] = Bm[tok * DSTATE + threadIdx.x];
        else if (threadIdx.x < 2 * DSTATE)
            sC[threadIdx.x - DSTATE] = Cm[tok * DSTATE + threadIdx.x - DSTATE];
        __syncthreads();
        const float dtv = dt[tok * DINNER + e];
        const float xv  = xc[tok * DINNER + e];
        const float xd  = xv * dtv;
        float y = 0.f;
#pragma unroll
        for (int s = 0; s < DSTATE; s++) {
            h[s] = h[s] * __expf(Ar[s] * dtv) + xd * sB[s];
            y += h[s] * sC[s];
        }
        const float z = xz[tok * (2 * DINNER) + DINNER + e];
        ybf[tok * DINNER + e] = f32_to_bf16((y + xv * de) * silu_f(z));
        __syncthreads();
    }
}

// ---------------------------------------------------------------- launcher
extern "C" void kernel_launch(void* const* d_in, const int* in_sizes, int n_in,
                              void* d_out, int out_size, void* d_ws, size_t ws_size,
                              hipStream_t stream) {
    const float* x        = (const float*)d_in[0];
    const float* norm_w   = (const float*)d_in[1];
    const float* norm_b   = (const float*)d_in[2];
    const float* in_projw = (const float*)d_in[3];
    const float* conv_w   = (const float*)d_in[4];
    const float* conv_b   = (const float*)d_in[5];
    const float* dt_w     = (const float*)d_in[6];
    const float* dt_b     = (const float*)d_in[7];
    const float* Bp_w     = (const float*)d_in[8];
    const float* Cp_w     = (const float*)d_in[9];
    const float* D_param  = (const float*)d_in[10];
    const float* A_log    = (const float*)d_in[11];
    const float* out_w    = (const float*)d_in[12];
    float* out = (float*)d_out;

    uint8_t* ws = (uint8_t*)d_ws;
    size_t off = 0;
    auto alloc = [&](size_t bytes) -> void* {
        void* p = ws + off;
        off = (off + bytes + 255) & ~(size_t)255;
        return p;
    };

    uint16_t* xn_bf   = (uint16_t*)alloc((size_t)NTOK * DMODEL * 2);
    uint16_t* w_in_bf = (uint16_t*)alloc((size_t)2 * DINNER * DMODEL * 2);
    uint16_t* w_dt_bf = (uint16_t*)alloc((size_t)DINNER * DINNER * 2);
    uint16_t* w_out_bf= (uint16_t*)alloc((size_t)DMODEL * DINNER * 2);
    uint16_t* w_B_bf  = (uint16_t*)alloc((size_t)DSTATE * DINNER * 2);
    uint16_t* w_C_bf  = (uint16_t*)alloc((size_t)DSTATE * DINNER * 2);
    float*    xz      = (float*)alloc((size_t)NTOK * 2 * DINNER * 4);
    float*    xc_f32  = (float*)alloc((size_t)NTOK * DINNER * 4);
    uint16_t* xc_bf   = (uint16_t*)alloc((size_t)NTOK * DINNER * 2);
    float*    dt_f32  = (float*)alloc((size_t)NTOK * DINNER * 4);
    float*    Bm      = (float*)alloc((size_t)NTOK * DSTATE * 4);
    float*    Cm      = (float*)alloc((size_t)NTOK * DSTATE * 4);
    uint16_t* y_bf    = (uint16_t*)alloc((size_t)NTOK * DINNER * 2);

    // weights fp32 -> bf16
    {
        int n;
        n = 2 * DINNER * DMODEL;
        convert_f32_bf16_kernel<<<(n + 255) / 256, 256, 0, stream>>>(in_projw, w_in_bf, n);
        n = DINNER * DINNER;
        convert_f32_bf16_kernel<<<(n + 255) / 256, 256, 0, stream>>>(dt_w, w_dt_bf, n);
        n = DMODEL * DINNER;
        convert_f32_bf16_kernel<<<(n + 255) / 256, 256, 0, stream>>>(out_w, w_out_bf, n);
        n = DSTATE * DINNER;
        convert_f32_bf16_kernel<<<(n + 255) / 256, 256, 0, stream>>>(Bp_w, w_B_bf, n);
        convert_f32_bf16_kernel<<<(n + 255) / 256, 256, 0, stream>>>(Cp_w, w_C_bf, n);
    }

    // LayerNorm -> bf16 xn
    layernorm_kernel<<<NTOK, 256, 0, stream>>>(x, norm_w, norm_b, xn_bf);

    // xz = xn @ in_proj_w^T : 8192x1024 * (4096x1024)^T
    // tiles = (8192/64)*(4096/64) = 8192 -> 1024 blocks
    wmma_gemm_bf16<4, 4, 0><<<1024, 256, 0, stream>>>(
        xn_bf, w_in_bf, nullptr, nullptr, xz, NTOK, 2 * DINNER, DMODEL);

    // depthwise conv + SiLU
    conv_silu_kernel<<<((size_t)NTOK * DINNER) / 256, 256, 0, stream>>>(
        xz, conv_w, conv_b, xc_f32, xc_bf);

    // dt = softplus(x_conv @ dt_w^T + dt_b) : tiles = 128*32 = 4096 -> 512 blocks
    wmma_gemm_bf16<4, 4, 1><<<512, 256, 0, stream>>>(
        xc_bf, w_dt_bf, dt_b, nullptr, dt_f32, NTOK, DINNER, DINNER);

    // Bm, Cm = x_conv @ {Bp,Cp}_w^T : tiles = 256*1 = 256 -> 32 blocks
    wmma_gemm_bf16<2, 1, 0><<<32, 256, 0, stream>>>(
        xc_bf, w_B_bf, nullptr, nullptr, Bm, NTOK, DSTATE, DINNER);
    wmma_gemm_bf16<2, 1, 0><<<32, 256, 0, stream>>>(
        xc_bf, w_C_bf, nullptr, nullptr, Cm, NTOK, DSTATE, DINNER);

    // selective scan + fused D-skip + SiLU(z) gate -> bf16 y
    scan_kernel<<<dim3(DINNER / 256, BSZ), 256, 0, stream>>>(
        dt_f32, xc_f32, Bm, Cm, xz, A_log, D_param, y_bf);

    // out = residual + y @ out_w^T : tiles = 128*16 = 2048 -> 256 blocks
    wmma_gemm_bf16<4, 4, 2><<<256, 256, 0, stream>>>(
        y_bf, w_out_bf, nullptr, x, out, NTOK, DMODEL, DINNER);
}
